// GATGraph_24343874633945
// MI455X (gfx1250) — matrix operand order repaired and checked by reading, not compiled, not run on hardware
//
#include <hip/hip_runtime.h>
#include <hip/hip_bf16.h>

// ---------------- CDNA5 WMMA types ----------------
typedef __attribute__((ext_vector_type(16))) __bf16 v16bf;
typedef __attribute__((ext_vector_type(8)))  float  v8f;

// Problem constants
constexpr int HD    = 128;   // hidden
constexpr int KNODE = 65;    // K+1 rows per batch element
constexpr int KDIFF = 32;
constexpr int NB    = 4096;  // batch

// Own trivially-copyable 16B structs (avoid HIP vector-type quirks in bit_cast)
struct alignas(16) U4 { unsigned x, y, z, w; };
struct alignas(16) F4 { float x, y, z, w; };
struct Frag32 { U4 lo, hi; };   // 32 bytes = 16 bf16 = one lane's WMMA operand

__device__ __forceinline__ unsigned short f2bf(float f) {
  unsigned u = __builtin_bit_cast(unsigned, f);
  u += 0x7FFFu + ((u >> 16) & 1u);          // round-to-nearest-even
  return (unsigned short)(u >> 16);
}
__device__ __forceinline__ float bf2f(unsigned short s) {
  unsigned u = ((unsigned)s) << 16;
  return __builtin_bit_cast(float, u);
}
__device__ __forceinline__ v8f v8zero() {
  v8f z = {0.f,0.f,0.f,0.f,0.f,0.f,0.f,0.f};
  return z;
}
__device__ __forceinline__ v8f wmma_bf16(v16bf a, v16bf b, v8f c) {
  // (neg_a, A, neg_b, B, c_mod, C, reuse_a, reuse_b)
  return __builtin_amdgcn_wmma_f32_16x16x32_bf16(false, a, false, b, (short)0, c, false, false);
}

// A-fragment (16x32 bf16) from LDS row-major buffer of width HD.
// CDNA5 16-bit A layout: lane%16 = M row; lane>=16 => K offset +8;
// lane holds K = kt*32 + hi*8 + {0..7} and kt*32 + 16 + hi*8 + {0..7}.
__device__ __forceinline__ v16bf ldsA(const unsigned short* buf, int mt, int kt,
                                      int laneM, int hi) {
  const unsigned short* row = buf + (mt * 16 + laneM) * HD + kt * 32 + hi * 8;
  Frag32 f;
  f.lo = *(const U4*)(row);        // K = base + 0..7
  f.hi = *(const U4*)(row + 16);   // K = base + 16..23
  return __builtin_bit_cast(v16bf, f);
}

// B-fragment (32x16 bf16) from fragment-major weights in workspace:
// block (kt,nt) stored as 32 lanes x 16 contiguous bf16; elem e<8: K=kt*32+hi*8+e,
// e>=8: K=kt*32+16+hi*8+(e-8); N = nt*16 + lane%16.
__device__ __forceinline__ v16bf ldB(const unsigned short* w, int ktC, int nt,
                                     int kt, int lane) {
  const unsigned short* p = w + (((nt * ktC + kt) * 32 + lane) << 4);
  Frag32 f;
  f.lo = *(const U4*)(p);
  f.hi = *(const U4*)(p + 8);
  return __builtin_bit_cast(v16bf, f);
}

// Epilogue: D tile (16x16 f32 in v8f) + bias -> relu -> bf16 into LDS.
// C/D layout: VGPR r holds row M = mt*16 + hi*8 + r, col N = nt*16 + lane%16.
__device__ __forceinline__ void store_tile(unsigned short* dst, v8f acc, int mt,
                                           int nt, const float* bias,
                                           int laneM, int hi) {
  int n = nt * 16 + laneM;
  float bv = bias[n];
#pragma unroll
  for (int r = 0; r < 8; ++r) {
    int m = mt * 16 + hi * 8 + r;
    float v = acc[r] + bv;
    v = v > 0.f ? v : 0.f;
    dst[m * HD + n] = f2bf(v);
  }
}

// ---------------- weight prep: f32 row-major -> bf16 fragment-major ----------------
__global__ void prep_weight(const float* __restrict__ W,
                            unsigned short* __restrict__ dst, int Krows) {
  int o = blockIdx.x * blockDim.x + threadIdx.x;
  int total = Krows * HD;
  if (o >= total) return;
  int ktC   = Krows >> 5;      // K tiles of 32
  int perNt = ktC << 9;        // ktC * 32 lanes * 16 elems
  int nt   = o / perNt;
  int rem  = o - nt * perNt;
  int kt   = rem >> 9;
  int rem2 = rem & 511;
  int lanei = rem2 >> 4;
  int e     = rem2 & 15;
  int k = kt * 32 + ((lanei >= 16) ? 8 : 0) + ((e >= 8) ? 16 + (e - 8) : e);
  int n = nt * 16 + (lanei & 15);
  dst[o] = f2bf(W[k * HD + n]);
}

// ---------------- fused node_pref + transform ----------------
// One workgroup (128 threads = 4 wave32) per (side, batch element).
__global__ __launch_bounds__(128) void gat_fused(
    const float* __restrict__ user_text, const float* __restrict__ item_text,
    const float* __restrict__ user_diff_rel, const float* __restrict__ item_diff_rel,
    const int* __restrict__ user_inds, const int* __restrict__ item_inds,
    const float* __restrict__ user_emb, const float* __restrict__ item_emb,
    const float* __restrict__ ne_b1, const float* __restrict__ ne_b2,
    const float* __restrict__ att_b1, const float* __restrict__ att_W2,
    const float* __restrict__ att_b2,
    const float* __restrict__ tr_W1, const float* __restrict__ tr_b1,
    const float* __restrict__ tr_W2, const float* __restrict__ tr_b2,
    const unsigned short* __restrict__ wfrag, float* __restrict__ out) {
  // LDS: 3 x 20KB bf16 tile buffers + small f32 scratch  (~63 KB total)
  __shared__ unsigned short bufA[80 * HD];  // staging for A chunks (width 128)
  __shared__ unsigned short buf1[80 * HD];  // t1, later h
  __shared__ unsigned short bufN[80 * HD];  // nodes (relu'd)
  __shared__ float s_this[HD];
  __shared__ float s_sc[64];
  __shared__ float s_u[2 * HD];
  __shared__ float s_y1[HD];
  __shared__ float s_red;

  const int tid  = threadIdx.x;
  const int wave = tid >> 5, lane = tid & 31;
  const int laneM = lane & 15, hi = lane >> 4;
  const int bx   = blockIdx.x;
  const int side = bx >> 12;          // 0 = user, 1 = item
  const int b    = bx & (NB - 1);

  const float* text = side ? item_text : user_text;
  const float* drel = side ? item_diff_rel : user_diff_rel;
  const int*   inds = (side ? item_inds : user_inds) + b * KNODE;
  const float* dtab = side ? user_emb : item_emb;   // diff table
  const float* stab = side ? item_emb : user_emb;   // same table

  const unsigned short* wNe1 = wfrag;          // 256x128, ktC=8
  const unsigned short* wNe2 = wfrag + 32768;  // 128x128, ktC=4
  const unsigned short* wAtt = wfrag + 49152;  // 384x128, ktC=12

  // ===== Stage 1: x=[text|emb] (65x256) @ ne_W1 -> relu -> t1 =====
  v8f acc[5][2];
#pragma unroll
  for (int mt = 0; mt < 5; ++mt) { acc[mt][0] = v8zero(); acc[mt][1] = v8zero(); }

  for (int chunk = 0; chunk < 2; ++chunk) {
    // stage 128-wide K-chunk of x into bufA (rows 65..79 zero-padded)
    for (int g = tid; g < 80 * 32; g += 128) {
      int r = g >> 5, c4 = (g & 31) << 2;
      F4 v = {0.f, 0.f, 0.f, 0.f};
      if (r < KNODE) {
        if (chunk == 0) {
          v = *(const F4*)(text + ((size_t)(b * KNODE + r)) * HD + c4);
        } else {
          const float* tab = (r < KDIFF) ? dtab : stab;
          v = *(const F4*)(tab + (size_t)inds[r] * HD + c4);
        }
      }
      unsigned short* d = bufA + r * HD + c4;
      d[0] = f2bf(v.x); d[1] = f2bf(v.y); d[2] = f2bf(v.z); d[3] = f2bf(v.w);
    }
    __syncthreads();
#pragma unroll
    for (int kt = 0; kt < 4; ++kt) {
      int ka = chunk * 4 + kt;
      v16bf b0 = ldB(wNe1, 8, wave * 2 + 0, ka, lane);
      v16bf b1 = ldB(wNe1, 8, wave * 2 + 1, ka, lane);
#pragma unroll
      for (int mt = 0; mt < 5; ++mt) {
        v16bf a = ldsA(bufA, mt, kt, laneM, hi);
        acc[mt][0] = wmma_bf16(a, b0, acc[mt][0]);
        acc[mt][1] = wmma_bf16(a, b1, acc[mt][1]);
      }
    }
    __syncthreads();
  }
#pragma unroll
  for (int mt = 0; mt < 5; ++mt) {
    store_tile(buf1, acc[mt][0], mt, wave * 2 + 0, ne_b1, laneM, hi);
    store_tile(buf1, acc[mt][1], mt, wave * 2 + 1, ne_b1, laneM, hi);
  }
  __syncthreads();

  // ===== Stage 2: t1 (65x128) @ ne_W2 -> relu -> nodes =====
#pragma unroll
  for (int mt = 0; mt < 5; ++mt) { acc[mt][0] = v8zero(); acc[mt][1] = v8zero(); }
#pragma unroll
  for (int kt = 0; kt < 4; ++kt) {
    v16bf b0 = ldB(wNe2, 4, wave * 2 + 0, kt, lane);
    v16bf b1 = ldB(wNe2, 4, wave * 2 + 1, kt, lane);
#pragma unroll
    for (int mt = 0; mt < 5; ++mt) {
      v16bf a = ldsA(buf1, mt, kt, laneM, hi);
      acc[mt][0] = wmma_bf16(a, b0, acc[mt][0]);
      acc[mt][1] = wmma_bf16(a, b1, acc[mt][1]);
    }
  }
#pragma unroll
  for (int mt = 0; mt < 5; ++mt) {
    store_tile(bufN, acc[mt][0], mt, wave * 2 + 0, ne_b2, laneM, hi);
    store_tile(bufN, acc[mt][1], mt, wave * 2 + 1, ne_b2, laneM, hi);
  }
  __syncthreads();
  s_this[tid] = bf2f(bufN[64 * HD + tid]);   // this = nodes[:, -1]
  __syncthreads();

  // ===== Attention: feat=[this | rel | ne] (64x384) @ att_W1 -> relu -> h =====
  v8f acc2[4][2];
#pragma unroll
  for (int mt = 0; mt < 4; ++mt) { acc2[mt][0] = v8zero(); acc2[mt][1] = v8zero(); }

  for (int chunk = 0; chunk < 3; ++chunk) {
    for (int g = tid; g < 64 * 32; g += 128) {
      int r = g >> 5, c4 = (g & 31) << 2;
      unsigned short* d = bufA + r * HD + c4;
      if (chunk == 0) {                       // broadcast `this`
        d[0] = f2bf(s_this[c4 + 0]); d[1] = f2bf(s_this[c4 + 1]);
        d[2] = f2bf(s_this[c4 + 2]); d[3] = f2bf(s_this[c4 + 3]);
      } else if (chunk == 1) {                // rel = [diff_rel | ne[32:]*this]
        if (r < KDIFF) {
          F4 v = *(const F4*)(drel + ((size_t)(b * KDIFF + r)) * HD + c4);
          d[0] = f2bf(v.x); d[1] = f2bf(v.y); d[2] = f2bf(v.z); d[3] = f2bf(v.w);
        } else {
          const unsigned short* nr = bufN + r * HD + c4;
          d[0] = f2bf(bf2f(nr[0]) * s_this[c4 + 0]);
          d[1] = f2bf(bf2f(nr[1]) * s_this[c4 + 1]);
          d[2] = f2bf(bf2f(nr[2]) * s_this[c4 + 2]);
          d[3] = f2bf(bf2f(nr[3]) * s_this[c4 + 3]);
        }
      } else {                                // ne
        const unsigned short* nr = bufN + r * HD + c4;
        d[0] = nr[0]; d[1] = nr[1]; d[2] = nr[2]; d[3] = nr[3];
      }
    }
    __syncthreads();
#pragma unroll
    for (int kt = 0; kt < 4; ++kt) {
      int ka = chunk * 4 + kt;
      v16bf b0 = ldB(wAtt, 12, wave * 2 + 0, ka, lane);
      v16bf b1 = ldB(wAtt, 12, wave * 2 + 1, ka, lane);
#pragma unroll
      for (int mt = 0; mt < 4; ++mt) {
        v16bf a = ldsA(bufA, mt, kt, laneM, hi);
        acc2[mt][0] = wmma_bf16(a, b0, acc2[mt][0]);
        acc2[mt][1] = wmma_bf16(a, b1, acc2[mt][1]);
      }
    }
    __syncthreads();
  }
#pragma unroll
  for (int mt = 0; mt < 4; ++mt) {            // h -> buf1 rows 0..63
    store_tile(buf1, acc2[mt][0], mt, wave * 2 + 0, att_b1, laneM, hi);
    store_tile(buf1, acc2[mt][1], mt, wave * 2 + 1, att_b1, laneM, hi);
  }
  __syncthreads();

  // ===== scores s = relu(h @ att_W2 + b2), softmax over k, agg =====
  if (tid < 64) {
    float sacc = att_b2[0];
    const unsigned short* hr = buf1 + tid * HD;
    for (int n = 0; n < HD; ++n) sacc += bf2f(hr[n]) * att_W2[n];
    s_sc[tid] = sacc > 0.f ? sacc : 0.f;
  }
  __syncthreads();
  if (tid == 0) {
    float m = s_sc[0];
    for (int i = 1; i < 64; ++i) m = fmaxf(m, s_sc[i]);
    s_red = m;
  }
  __syncthreads();
  if (tid < 64) s_sc[tid] = __expf(s_sc[tid] - s_red);
  __syncthreads();
  if (tid == 0) {
    float sum = 0.f;
    for (int i = 0; i < 64; ++i) sum += s_sc[i];
    s_red = sum;
  }
  __syncthreads();
  if (tid < 64) s_sc[tid] = s_sc[tid] / s_red;
  __syncthreads();

  {                                           // agg = att . ne ; u = [this|agg]
    float a = 0.f;
    for (int m = 0; m < 64; ++m) a += s_sc[m] * bf2f(bufN[m * HD + tid]);
    s_u[tid]      = s_this[tid];
    s_u[HD + tid] = a;
  }
  __syncthreads();

  // ===== transform: relu(relu(u @ tr_W1 + b1) @ tr_W2 + b2) (f32 VALU) =====
  {
    float y = tr_b1[tid];
    for (int i = 0; i < 2 * HD; ++i) y += s_u[i] * tr_W1[i * HD + tid];
    s_y1[tid] = y > 0.f ? y : 0.f;
  }
  __syncthreads();
  {
    float o = tr_b2[tid];
    for (int i = 0; i < HD; ++i) o += s_y1[i] * tr_W2[i * HD + tid];
    o = o > 0.f ? o : 0.f;
    out[((size_t)side * NB + b) * HD + tid] = o;
  }
}

extern "C" void kernel_launch(void* const* d_in, const int* in_sizes, int n_in,
                              void* d_out, int out_size, void* d_ws, size_t ws_size,
                              hipStream_t stream) {
  const float* user_text     = (const float*)d_in[0];
  const float* item_text     = (const float*)d_in[1];
  const float* user_diff_rel = (const float*)d_in[2];
  const float* item_diff_rel = (const float*)d_in[3];
  const int*   user_inds     = (const int*)d_in[4];
  const int*   item_inds     = (const int*)d_in[5];
  const float* user_emb      = (const float*)d_in[6];
  const float* item_emb      = (const float*)d_in[7];
  const float* ne_W1 = (const float*)d_in[8];
  const float* ne_b1 = (const float*)d_in[9];
  const float* ne_W2 = (const float*)d_in[10];
  const float* ne_b2 = (const float*)d_in[11];
  const float* att_W1 = (const float*)d_in[12];
  const float* att_b1 = (const float*)d_in[13];
  const float* att_W2 = (const float*)d_in[14];
  const float* att_b2 = (const float*)d_in[15];
  const float* tr_W1 = (const float*)d_in[16];
  const float* tr_b1 = (const float*)d_in[17];
  const float* tr_W2 = (const float*)d_in[18];
  const float* tr_b2 = (const float*)d_in[19];

  unsigned short* ws = (unsigned short*)d_ws;
  // bf16 fragment-major weights: ne_W1 [0,32768), ne_W2 [32768,49152), att_W1 [49152,98304)
  prep_weight<<<(256 * HD + 255) / 256, 256, 0, stream>>>(ne_W1, ws, 256);
  prep_weight<<<(128 * HD + 255) / 256, 256, 0, stream>>>(ne_W2, ws + 32768, 128);
  prep_weight<<<(384 * HD + 255) / 256, 256, 0, stream>>>(att_W1, ws + 49152, 384);

  gat_fused<<<2 * NB, 128, 0, stream>>>(
      user_text, item_text, user_diff_rel, item_diff_rel, user_inds, item_inds,
      user_emb, item_emb, ne_b1, ne_b2, att_b1, att_W2, att_b2, tr_W1, tr_b1,
      tr_W2, tr_b2, (const unsigned short*)d_ws, (float*)d_out);
}
